// TransformerBlock_87213605913030
// MI455X (gfx1250) — matrix operand order repaired and checked by reading, not compiled, not run on hardware
//
#include <hip/hip_runtime.h>
#include <math.h>

typedef __attribute__((ext_vector_type(16))) _Float16 v16h;
typedef __attribute__((ext_vector_type(8)))  float    v8f;

// Problem constants (match reference)
#define CDIM   16
#define DKDIM  8
#define SHDIM  16
#define NBASIS 10
#define KTOT   256   // CDIM * SHDIM
#define COLS   128   // 16 m * 8 o

// ---------------------------------------------------------------------------
// Pack Wk2/Wv2 [16,2048] f32 -> WMMA-B fragment-ordered f16 [8 ct][8 kb][32 lane][16 half]
// B layout (v_wmma_*_16x16x32, K=32): lanes 0-15 hold K={0..7,16..23} (2 per VGPR),
// lanes 16-31 hold K={8..15,24..31}; lane%16 = N column.
// Bmat[cj][m*8+o] = W2[m][cj*8+o]
// ---------------------------------------------------------------------------
__global__ void pack_w2_kernel(const float* __restrict__ Wk2,
                               const float* __restrict__ Wv2,
                               _Float16* __restrict__ BK,
                               _Float16* __restrict__ BV) {
  int idx = blockIdx.x * blockDim.x + threadIdx.x;
  if (idx >= 2 * 32768) return;
  int sel = idx >> 15;
  int s   = idx & 32767;
  int i   = s & 15;          // half index within lane
  int L   = (s >> 4) & 31;   // lane
  int kb  = (s >> 9) & 7;    // K-block (32 K each)
  int ct  = (s >> 12) & 7;   // column tile (16 cols each)
  int col = ct * 16 + (L & 15);
  int m = col >> 3, o = col & 7;
  int cj = kb * 32 + ((L < 16) ? 0 : 8) + i + ((i >= 8) ? 8 : 0);
  const float* W2 = sel ? Wv2 : Wk2;
  _Float16*    B  = sel ? BV  : BK;
  B[s] = (_Float16)W2[m * 2048 + cj * 8 + o];
}

// q = x @ Wq  ([N,16] x [16,8])
__global__ void q_kernel(const float* __restrict__ x, const float* __restrict__ Wq,
                         float* __restrict__ q, int N) {
  int n = blockIdx.x * blockDim.x + threadIdx.x;
  if (n >= N) return;
  float xr[CDIM];
#pragma unroll
  for (int c = 0; c < CDIM; ++c) xr[c] = x[n * CDIM + c];
#pragma unroll
  for (int o = 0; o < DKDIM; ++o) {
    float acc = 0.f;
#pragma unroll
    for (int c = 0; c < CDIM; ++c) acc += xr[c] * Wq[c * DKDIM + o];
    q[n * DKDIM + o] = acc;
  }
}

__global__ void zero_kernel(float* __restrict__ out, int nout,
                            float* __restrict__ z, int nz) {
  int i = blockIdx.x * blockDim.x + threadIdx.x;
  if (i < nout) out[i] = 0.f;
  if (i < nz)   z[i]   = 0.f;
}

// ---------------------------------------------------------------------------
// Fused per-edge kernel: 16 edges per block, 256 threads (8 wave32).
// Wave w owns output columns [w*16, w*16+16) of the [16,128] GEMM tile for
// both K and V generators; results contracted with h via LDS atomics.
// ---------------------------------------------------------------------------
__global__ void edge_main_kernel(const float* __restrict__ pos,
                                 const float* __restrict__ x,
                                 const float* __restrict__ Wk1,
                                 const float* __restrict__ Wv1,
                                 const _Float16* __restrict__ BK,
                                 const _Float16* __restrict__ BV,
                                 const float* __restrict__ qbuf,
                                 const int* __restrict__ edge_src,
                                 const int* __restrict__ edge_dst,
                                 float* __restrict__ expv_out,
                                 float* __restrict__ v_out,
                                 float* __restrict__ zbuf,
                                 int E) {
  __shared__ v16h  GaV[8 * 32];        // A fragments: [kb][lane] 16 halves each (8KB)
  __shared__ float sh_s[16][SHDIM];
  __shared__ float xs_s[16][CDIM];
  __shared__ float hk_s[16][16];
  __shared__ float hv_s[16][16];
  __shared__ float kacc[16][DKDIM];
  __shared__ float vacc[16][DKDIM];
  __shared__ float cut_s[16];
  __shared__ int   dst_s[16];

  const int tid = threadIdx.x;
  if (tid < 128) { ((float*)kacc)[tid] = 0.f; ((float*)vacc)[tid] = 0.f; }

  // ---- per-edge scalar pipeline (threads 0..15, one edge each) ----
  if (tid < 16) {
    const int ge = blockIdx.x * 16 + tid;
    if (ge < E) {
      const int src = edge_src[ge];
      const int dst = edge_dst[ge];
      dst_s[tid] = dst;
      const float vx = pos[src * 3 + 0] - pos[dst * 3 + 0];
      const float vy = pos[src * 3 + 1] - pos[dst * 3 + 1];
      const float vz = pos[src * 3 + 2] - pos[dst * 3 + 2];
      const float d2 = vx * vx + vy * vy + vz * vz;
      const float r  = sqrtf(d2 + 1e-18f);

      // smooth_finite radial basis
      float emb[NBASIS];
      const float step = 3.5f / 11.0f;
      const float embScale = 1.14136f * 7.3890561f * 3.1622777f; // 1.14136*e^2*sqrt(10)
#pragma unroll
      for (int b = 0; b < NBASIS; ++b) {
        float u = (r - (float)(b + 1) * step) / step;
        float y = 0.f;
        if (fabsf(u) < 1.f) y = expf(-1.f / (1.f - u * u));
        emb[b] = embScale * y;
      }
      const float t = 10.f * (1.f - r * (1.f / 3.5f));
      cut_s[tid] = (t > 0.f) ? expf(-1.f / t) : 0.f;

      // real spherical harmonics l=0..3 (component norm)
      float rn = sqrtf(d2); rn = (rn > 1e-9f) ? rn : 1e-9f;
      const float ux = vx / rn, uy = vy / rn, uz = vz / rn;
      const float s3 = 1.7320508f, s15 = 3.8729833f;
      const float s5h = 1.1180340f, s15h = 1.9364917f;
      const float c358 = 2.0916500f, c218 = 1.6201852f;
      const float s105 = 10.246951f, s105h = 5.1234756f, s7h = 1.3228757f;
      const float xx = ux * ux, yy = uy * uy, zz = uz * uz;
      sh_s[tid][0]  = 1.f;
      sh_s[tid][1]  = s3 * ux;
      sh_s[tid][2]  = s3 * uy;
      sh_s[tid][3]  = s3 * uz;
      sh_s[tid][4]  = s15 * ux * uy;
      sh_s[tid][5]  = s15 * uy * uz;
      sh_s[tid][6]  = s5h * (3.f * zz - 1.f);
      sh_s[tid][7]  = s15 * ux * uz;
      sh_s[tid][8]  = s15h * (xx - yy);
      sh_s[tid][9]  = c358 * uy * (3.f * xx - yy);
      sh_s[tid][10] = s105 * ux * uy * uz;
      sh_s[tid][11] = c218 * uy * (5.f * zz - 1.f);
      sh_s[tid][12] = s7h * uz * (5.f * zz - 3.f);
      sh_s[tid][13] = c218 * ux * (5.f * zz - 1.f);
      sh_s[tid][14] = s105h * uz * (xx - yy);
      sh_s[tid][15] = c358 * ux * (xx - 3.f * yy);

#pragma unroll
      for (int c = 0; c < CDIM; ++c) xs_s[tid][c] = x[src * CDIM + c];

      // first MLP layer: h = silu(emb @ W1 / sqrt(10))
      const float inv_s10 = 0.31622777f;
#pragma unroll
      for (int m = 0; m < 16; ++m) {
        float ak = 0.f, av = 0.f;
#pragma unroll
        for (int b = 0; b < NBASIS; ++b) {
          ak += emb[b] * Wk1[b * 16 + m];
          av += emb[b] * Wv1[b * 16 + m];
        }
        ak *= inv_s10; av *= inv_s10;
        hk_s[tid][m] = ak / (1.f + expf(-ak));
        hv_s[tid][m] = av / (1.f + expf(-av));
      }
    } else {
      dst_s[tid] = 0; cut_s[tid] = 0.f;
#pragma unroll
      for (int c = 0; c < 16; ++c) {
        sh_s[tid][c] = 0.f; xs_s[tid][c] = 0.f;
        hk_s[tid][c] = 0.f; hv_s[tid][c] = 0.f;
      }
    }
  }
  __syncthreads();

  // ---- build A fragments (G = xs outer sh) directly in WMMA A layout ----
  // A layout: lanes 0-15 row M=lane, K = {0..7, 16..23}; lanes 16-31 K = {8..15, 24..31}
  {
    const int kb = tid >> 5, lane = tid & 31;
    const int e = lane & 15;
    const int base = (lane < 16) ? 0 : 8;
    v16h tmp = {};
#pragma unroll
    for (int i = 0; i < 16; ++i) {
      int ko = kb * 32 + base + i + ((i >= 8) ? 8 : 0); // global K = c*16+j
      int c = ko >> 4, j = ko & 15;
      tmp[i] = (_Float16)(xs_s[e][c] * sh_s[e][j]);
    }
    GaV[kb * 32 + lane] = tmp;
  }
  __syncthreads();

  // ---- WMMA GEMM: [16 edges x 256] x [256 x 16 cols per wave], K and V ----
  const int w = tid >> 5;
  const int lane = tid & 31;
  const v16h* __restrict__ BKv = (const v16h*)BK;
  const v16h* __restrict__ BVv = (const v16h*)BV;
  v8f ck = {}; v8f cv = {};
#pragma unroll
  for (int kb = 0; kb < 8; ++kb) {
    v16h a  = GaV[kb * 32 + lane];
    v16h bk = BKv[(w * 8 + kb) * 32 + lane];
    v16h bv = BVv[(w * 8 + kb) * 32 + lane];
    ck = __builtin_amdgcn_wmma_f32_16x16x32_f16(false, a, false, bk, (short)0, ck, false, false);
    cv = __builtin_amdgcn_wmma_f32_16x16x32_f16(false, a, false, bv, (short)0, cv, false, false);
  }

  // ---- contract with h over m: D layout lane->N col, VGPR r -> M row ----
  {
    const int col = w * 16 + (lane & 15);
    const int m = col >> 3, o = col & 7;
    const int eb = (lane < 16) ? 0 : 8;
#pragma unroll
    for (int r = 0; r < 8; ++r) {
      const int e = eb + r;
      atomicAdd(&kacc[e][o], hk_s[e][m] * ck[r]);
      atomicAdd(&vacc[e][o], hv_s[e][m] * cv[r]);
    }
  }
  __syncthreads();

  // ---- logits, expv, segment-sum denominator ----
  if (tid < 16) {
    const int ge = blockIdx.x * 16 + tid;
    if (ge < E) {
      const int dst = dst_s[tid];
      const float inv64 = 1.0f / 64.0f;       // 1/(sqrt(16)*sqrt(256))
      float logit = 0.f;
#pragma unroll
      for (int o = 0; o < DKDIM; ++o)
        logit += qbuf[dst * DKDIM + o] * (kacc[tid][o] * inv64);
      logit *= 0.35355339f;                   // 1/sqrt(8)
      const float ev = cut_s[tid] * expf(logit);
      expv_out[ge] = ev;
      atomicAdd(&zbuf[dst], ev);
#pragma unroll
      for (int o = 0; o < DKDIM; ++o)
        v_out[ge * DKDIM + o] = vacc[tid][o] * inv64;
    }
  }
}

// ---------------------------------------------------------------------------
// alpha = expv / z[dst]; out[dst] += sqrt(relu(alpha)) * v
// ---------------------------------------------------------------------------
__global__ void finalize_kernel(const float* __restrict__ expv,
                                const float* __restrict__ vbuf,
                                const float* __restrict__ zbuf,
                                const int* __restrict__ edge_dst,
                                float* __restrict__ out, int E) {
  int e = blockIdx.x * blockDim.x + threadIdx.x;
  if (e >= E) return;
  const int dst = edge_dst[e];
  float z = zbuf[dst];
  if (z == 0.0f) z = 1.0f;
  float a = expv[e] / z;
  a = (a > 0.f) ? a : 0.f;
  const float wgt = sqrtf(a);
#pragma unroll
  for (int o = 0; o < DKDIM; ++o)
    atomicAdd(&out[dst * DKDIM + o], wgt * vbuf[e * DKDIM + o]);
}

extern "C" void kernel_launch(void* const* d_in, const int* in_sizes, int n_in,
                              void* d_out, int out_size, void* d_ws, size_t ws_size,
                              hipStream_t stream) {
  const float* pos  = (const float*)d_in[0];
  const float* x    = (const float*)d_in[1];
  const float* Wq   = (const float*)d_in[2];
  const float* Wk1  = (const float*)d_in[3];
  const float* Wk2  = (const float*)d_in[4];
  const float* Wv1  = (const float*)d_in[5];
  const float* Wv2  = (const float*)d_in[6];
  const int*   esrc = (const int*)d_in[7];
  const int*   edst = (const int*)d_in[8];
  (void)esrc; (void)n_in; (void)ws_size;

  const int N = in_sizes[1] / CDIM;
  const int E = in_sizes[7];
  float* out = (float*)d_out;

  // workspace carve-up
  char* base = (char*)d_ws;
  size_t off = 0;
  _Float16* BK = (_Float16*)(base + off); off += 32768 * sizeof(_Float16);
  _Float16* BV = (_Float16*)(base + off); off += 32768 * sizeof(_Float16);
  float* qbuf = (float*)(base + off); off += (size_t)N * DKDIM * sizeof(float);
  off = (off + 255) & ~(size_t)255;
  float* zbuf = (float*)(base + off); off += (size_t)N * sizeof(float);
  off = (off + 255) & ~(size_t)255;
  float* expv = (float*)(base + off); off += (size_t)E * sizeof(float);
  off = (off + 255) & ~(size_t)255;
  float* vbuf = (float*)(base + off); off += (size_t)E * DKDIM * sizeof(float);

  // 1) pack W2 matrices into WMMA-B fragment order (f16)
  pack_w2_kernel<<<(2 * 32768 + 255) / 256, 256, 0, stream>>>(Wk2, Wv2, BK, BV);
  // 2) q = x @ Wq
  q_kernel<<<(N + 255) / 256, 256, 0, stream>>>(x, Wq, qbuf, N);
  // 3) zero accumulators (d_out is poisoned by harness)
  zero_kernel<<<((N * DKDIM) + 255) / 256, 256, 0, stream>>>(out, N * DKDIM, zbuf, N);
  // 4) fused per-edge WMMA kernel
  edge_main_kernel<<<(E + 15) / 16, 256, 0, stream>>>(
      pos, x, Wk1, Wv1, BK, BV, qbuf, esrc, edst, expv, vbuf, zbuf, E);
  // 5) softmax normalize + scatter
  finalize_kernel<<<(E + 255) / 256, 256, 0, stream>>>(expv, vbuf, zbuf, edst, out, E);
  (void)out_size;
}
